// CrossScaleAttention_5325759447374
// MI455X (gfx1250) — matrix operand order repaired and checked by reading, not compiled, not run on hardware
//
#include <hip/hip_runtime.h>
#include <cstdint>

typedef _Float16 half_t;
typedef __attribute__((ext_vector_type(8)))  _Float16 v8h;
typedef __attribute__((ext_vector_type(16))) _Float16 v16h;
typedef __attribute__((ext_vector_type(8)))  float    v8f;

union V16 { v16h v; v8h h[2]; };

// ---- problem constants ----
#define NB   4
#define CIN  64
#define C2C  32
#define HH   60
#define WW   60
#define HWX  3600
#define LL   400     // number of patches (20x20)
#define LPAD 416     // LL padded to multiple of 32 for WMMA K
#define K1   288     // C2C*3*3 correlation K
#define HP_E 66      // embed padded (pad 3 each side)
#define HP_M 62      // match padded (pad 1 each side)
#define HP_R 22      // ref padded (pad 1 each side)
#define HOUT 180
#define MT_C 25      // corr M tiles (400/16)
#define NT_X 225     // N tiles (3600/16)
#define MG_C 13      // ceil(25/2)
#define NG_X 113     // ceil(225/2)

__device__ __forceinline__ float prelu(float y, float a) { return y >= 0.f ? y : a * y; }

// ---------------------------------------------------------------------------
// 1x1 conv + PReLU into a zero-padded NCHW buffer (pad on H and W).
// x: [NB][CIN][60][60], Wt: [Cout][CIN], outp: [NB][Cout][Hp][Hp]
__global__ void k_conv1x1_pad(const float* __restrict__ x, const float* __restrict__ Wt,
                              const float* __restrict__ bias, const float* __restrict__ alpha,
                              float* __restrict__ outp, int Cout, int pad, int Hp) {
    int idx = blockIdx.x * blockDim.x + threadIdx.x;
    int total = NB * Cout * Hp * Hp;
    if (idx >= total) return;
    int wp = idx % Hp; int t = idx / Hp;
    int hp = t % Hp;   t /= Hp;
    int c  = t % Cout; int b = t / Cout;
    int h = hp - pad, w = wp - pad;
    float v = 0.f;
    if (h >= 0 && h < HH && w >= 0 && w < WW) {
        const float* xb = x + ((size_t)b * CIN) * (HH * WW) + h * WW + w;
        const float* wr = Wt + c * CIN;
        float acc = bias[c];
        #pragma unroll 8
        for (int ci = 0; ci < CIN; ++ci) acc += wr[ci] * xb[ci * (HH * WW)];
        v = prelu(acc, alpha[0]);
    }
    outp[idx] = v;
}

// ---------------------------------------------------------------------------
// ref = PReLU(Wm2 @ bilinear_down(x,20,20) + bm2); the bilinear coords
// (u+0.5)*3-0.5 = 3u+1 are exact integers -> point sample at (3u+1, 3v+1).
// Written zero-padded: refp [NB][C2C][22][22].
__global__ void k_ref_pad(const float* __restrict__ x, const float* __restrict__ Wm2,
                          const float* __restrict__ bm2, const float* __restrict__ am2,
                          float* __restrict__ refp) {
    int idx = blockIdx.x * blockDim.x + threadIdx.x;
    int total = NB * C2C * HP_R * HP_R;
    if (idx >= total) return;
    int wp = idx % HP_R; int t = idx / HP_R;
    int hp = t % HP_R;   t /= HP_R;
    int c  = t % C2C;    int b = t / C2C;
    int u = hp - 1, v = wp - 1;
    float r = 0.f;
    if (u >= 0 && u < 20 && v >= 0 && v < 20) {
        int sh = 3 * u + 1, sw = 3 * v + 1;
        const float* xb = x + ((size_t)b * CIN) * (HH * WW) + sh * WW + sw;
        const float* wr = Wm2 + c * CIN;
        float acc = bm2[c];
        #pragma unroll 8
        for (int ci = 0; ci < CIN; ++ci) acc += wr[ci] * xb[ci * (HH * WW)];
        r = prelu(acc, am2[0]);
    }
    refp[idx] = r;
}

// ---------------------------------------------------------------------------
// 3x3 patches of refp, L2-normalized over 288 elems -> wn [NB][LL][K1] f16.
// One 128-thread block per (b,l).
__global__ void k_patchnorm(const float* __restrict__ refp, half_t* __restrict__ wn) {
    int bl = blockIdx.x;
    int b = bl / LL, l = bl % LL;
    int lh = l / 20, lw = l % 20;
    int tid = threadIdx.x;
    float vals[3];
    float ss = 0.f;
    #pragma unroll
    for (int i = 0; i < 3; ++i) {
        int k = tid + i * 128;
        float v = 0.f;
        if (k < K1) {
            int c = k / 9, r = k % 9;
            int kh = r / 3, kw = r % 3;
            v = refp[(((size_t)b * C2C + c) * HP_R + (lh + kh)) * HP_R + (lw + kw)];
        }
        vals[i] = v;
        ss += v * v;
    }
    __shared__ float red[128];
    red[tid] = ss;
    __syncthreads();
    for (int s = 64; s > 0; s >>= 1) {
        if (tid < s) red[tid] += red[tid + s];
        __syncthreads();
    }
    float inv = 1.f / fmaxf(sqrtf(red[0]), 1e-4f);
    #pragma unroll
    for (int i = 0; i < 3; ++i) {
        int k = tid + i * 128;
        if (k < K1) wn[((size_t)b * LL + l) * K1 + k] = (half_t)(vals[i] * inv);
    }
}

// ---------------------------------------------------------------------------
// im2col of padded match, K-contiguous ("B transposed"): bcol [NB][HWX][K1] f16
__global__ void k_im2col(const float* __restrict__ xp, half_t* __restrict__ bcol) {
    int idx = blockIdx.x * blockDim.x + threadIdx.x;
    int total = NB * HWX * K1;
    if (idx >= total) return;
    int k  = idx % K1; int t = idx / K1;
    int hw = t % HWX;  int b = t / HWX;
    int c = k / 9, r = k % 9;
    int kh = r / 3, kw = r % 3;
    int h = hw / WW, w = hw % WW;
    bcol[idx] = (half_t)xp[(((size_t)b * C2C + c) * HP_M + (h + kh)) * HP_M + (w + kw)];
}

// ---------------------------------------------------------------------------
// Reorganize 9x9-stride-3 patches of padded embed_w as tconv GEMM A matrices:
// rawT [NB][9(kh)][9(kw)][CIN][LPAD] f16 (zero for l >= LL).
__global__ void k_rawT(const float* __restrict__ embp, half_t* __restrict__ rawT) {
    int idx = blockIdx.x * blockDim.x + threadIdx.x;
    int total = NB * 81 * CIN * LPAD;
    if (idx >= total) return;
    int lp = idx % LPAD; int t = idx / LPAD;
    int c  = t % CIN;    t /= CIN;
    int kw = t % 9;      t /= 9;
    int kh = t % 9;      int b = t / 9;
    float v = 0.f;
    if (lp < LL) {
        int lh = lp / 20, lw = lp % 20;
        v = embp[(((size_t)b * CIN + c) * HP_E + (3 * lh + kh)) * HP_E + (3 * lw + kw)];
    }
    rawT[idx] = (half_t)v;
}

// ---------------------------------------------------------------------------
// Correlation GEMM: yraw[b][l][hw] = sum_k wn[b][l][k] * bcol[b][hw][k]
// 2x2 register-blocked: each wave owns a 32x32 block (four 16x16 WMMA tiles),
// reusing each A/B fragment twice -> 2 b128 loads per WMMA instead of 4.
// Edge groups clamp to the last tile (duplicate work, identical writes).
__global__ void k_gemm_corr(const half_t* __restrict__ wn, const half_t* __restrict__ bcol,
                            float* __restrict__ yraw) {
    const int wavesTotal = NB * MG_C * NG_X;
    int wid = blockIdx.x * (blockDim.x >> 5) + (threadIdx.x >> 5);
    if (wid >= wavesTotal) return;               // wave-uniform: EXEC stays full
    int lane = threadIdx.x & 31;
    int ng = wid % NG_X; int t = wid / NG_X;
    int mg = t % MG_C;   int b = t / MG_C;
    int l16 = lane & 15;
    int kA = (lane < 16) ? 0 : 16;
    int kB = (lane < 16) ? 0 : 16;

    int mt[2], nt[2];
    #pragma unroll
    for (int i = 0; i < 2; ++i) {
        mt[i] = min(mg * 2 + i, MT_C - 1);
        nt[i] = min(ng * 2 + i, NT_X - 1);
    }
    const half_t* Arow[2];
    const half_t* Brow[2];
    #pragma unroll
    for (int i = 0; i < 2; ++i) {
        Arow[i] = wn   + ((size_t)b * LL  + (mt[i] * 16 + l16)) * K1;
        Brow[i] = bcol + ((size_t)b * HWX + (nt[i] * 16 + l16)) * K1;
    }

    v8f acc[2][2] = {};
    #pragma unroll
    for (int kb = 0; kb < K1; kb += 32) {
        V16 a[2];
        #pragma unroll
        for (int i = 0; i < 2; ++i) {
            a[i].v = *(const v16h*)(Arow[i] + kb + kA);
        }
        #pragma unroll
        for (int j = 0; j < 2; ++j) {
            v16h bf = *(const v16h*)(Brow[j] + kb + kB);
            #pragma unroll
            for (int i = 0; i < 2; ++i)
                acc[i][j] = __builtin_amdgcn_wmma_f32_16x16x32_f16(false, a[i].v, false, bf,
                                                                   (short)0, acc[i][j],
                                                                   false, false);
        }
    }

    float* C = yraw + (size_t)b * LL * HWX;
    int rhalf = (lane < 16) ? 0 : 8;
    #pragma unroll
    for (int i = 0; i < 2; ++i) {
        int rbase = mt[i] * 16 + rhalf;
        #pragma unroll
        for (int j = 0; j < 2; ++j) {
            int n = nt[j] * 16 + l16;
            #pragma unroll
            for (int r = 0; r < 8; ++r)
                C[(size_t)(rbase + r) * HWX + n] = acc[i][j][r];
        }
    }
}

// ---------------------------------------------------------------------------
__global__ void k_zero32(uint32_t* __restrict__ p, int n) {
    int i = blockIdx.x * blockDim.x + threadIdx.x;
    if (i < n) p[i] = 0u;
}

// softmax over L (x10 scale), written l-fastest into padded yp [NB][62][62][LPAD] f16
__global__ void k_softmax(const float* __restrict__ yraw, half_t* __restrict__ yp) {
    int idx = blockIdx.x * blockDim.x + threadIdx.x;
    if (idx >= NB * HWX) return;
    int hw = idx % HWX, b = idx / HWX;
    int h = hw / WW, w = hw % WW;
    const float* col = yraw + (size_t)b * LL * HWX + hw;
    float mx = -1e30f;
    for (int l = 0; l < LL; ++l) mx = fmaxf(mx, col[(size_t)l * HWX] * 10.f);
    float s = 0.f;
    for (int l = 0; l < LL; ++l) s += __expf(col[(size_t)l * HWX] * 10.f - mx);
    float inv = 1.f / s;
    half_t* dst = yp + (((size_t)b * HP_M + (h + 1)) * HP_M + (w + 1)) * LPAD;
    for (int l = 0; l < LL; ++l)
        dst[l] = (half_t)(__expf(col[(size_t)l * HWX] * 10.f - mx) * inv);
}

// ---------------------------------------------------------------------------
// Transposed-conv as 9 phase GEMMs, each accumulating 9 shifted K=416 GEMMs:
// out[b][c][3u+p][3v+q] = (1/6) * sum_{dh,dw} sum_l
//     rawT[b][p+3(2-dh)][q+3(2-dw)][c][l] * yp[b][u+dh][v+dw][l]
// 2x2 register-blocked (M tiles 4 -> 2 groups exact; N edge clamped).
__global__ void k_gemm_tconv(const half_t* __restrict__ rawT, const half_t* __restrict__ yp,
                             float* __restrict__ out) {
    const int wavesTotal = NB * 3 * 3 * 2 * NG_X;   // 8136
    int wid = blockIdx.x * (blockDim.x >> 5) + (threadIdx.x >> 5);
    if (wid >= wavesTotal) return;                  // wave-uniform
    int lane = threadIdx.x & 31;
    int ng = wid % NG_X; int t = wid / NG_X;
    int mg = t % 2;      t /= 2;
    int q  = t % 3;      t /= 3;
    int p  = t % 3;      int b = t / 3;
    int l16 = lane & 15;
    int kA = (lane < 16) ? 0 : 16;
    int kB = (lane < 16) ? 0 : 16;

    int mt[2], nt[2], u[2], v[2];
    #pragma unroll
    for (int i = 0; i < 2; ++i) {
        mt[i] = mg * 2 + i;                         // 4 M tiles, exact
        nt[i] = min(ng * 2 + i, NT_X - 1);
        int n = nt[i] * 16 + l16;
        u[i] = n / WW;
        v[i] = n % WW;
    }

    v8f acc[2][2] = {};
    for (int dh = 0; dh < 3; ++dh) {
        for (int dw = 0; dw < 3; ++dw) {
            int kh = p + 3 * (2 - dh);
            int kw = q + 3 * (2 - dw);
            const half_t* Abase = rawT + (((size_t)b * 9 + kh) * 9 + kw) * CIN * LPAD;
            const half_t* Arow[2];
            const half_t* Brow[2];
            #pragma unroll
            for (int i = 0; i < 2; ++i) {
                Arow[i] = Abase + (size_t)(mt[i] * 16 + l16) * LPAD;
                Brow[i] = yp + (((size_t)b * HP_M + (u[i] + dh)) * HP_M + (v[i] + dw)) * LPAD;
            }
            #pragma unroll
            for (int kb = 0; kb < LPAD; kb += 32) {
                V16 a[2];
                #pragma unroll
                for (int i = 0; i < 2; ++i) {
                    a[i].v = *(const v16h*)(Arow[i] + kb + kA);
                }
                #pragma unroll
                for (int j = 0; j < 2; ++j) {
                    v16h bf = *(const v16h*)(Brow[j] + kb + kB);
                    #pragma unroll
                    for (int i = 0; i < 2; ++i)
                        acc[i][j] = __builtin_amdgcn_wmma_f32_16x16x32_f16(false, a[i].v, false, bf,
                                                                           (short)0, acc[i][j],
                                                                           false, false);
                }
            }
        }
    }

    int rhalf = (lane < 16) ? 0 : 8;
    #pragma unroll
    for (int i = 0; i < 2; ++i) {
        int rbase = mt[i] * 16 + rhalf;
        #pragma unroll
        for (int j = 0; j < 2; ++j) {
            int oh = 3 * u[j] + p, ow = 3 * v[j] + q;
            float* ob = out + (((size_t)b * CIN + rbase) * HOUT + oh) * HOUT + ow;
            #pragma unroll
            for (int r = 0; r < 8; ++r)
                ob[(size_t)r * HOUT * HOUT] = acc[i][j][r] * (1.f / 6.f);
        }
    }
}

// ---------------------------------------------------------------------------
extern "C" void kernel_launch(void* const* d_in, const int* in_sizes, int n_in,
                              void* d_out, int out_size, void* d_ws, size_t ws_size,
                              hipStream_t stream) {
    const float* x   = (const float*)d_in[0];
    const float* Wa  = (const float*)d_in[1];
    const float* ba  = (const float*)d_in[2];
    const float* aa  = (const float*)d_in[3];
    const float* Wm1 = (const float*)d_in[4];
    const float* bm1 = (const float*)d_in[5];
    const float* am1 = (const float*)d_in[6];
    const float* Wm2 = (const float*)d_in[7];
    const float* bm2 = (const float*)d_in[8];
    const float* am2 = (const float*)d_in[9];
    float* out = (float*)d_out;

    char* ws = (char*)d_ws;
    size_t off = 0;
    auto carve = [&](size_t bytes) -> char* {
        char* pp = ws + off;
        off = (off + bytes + 255) & ~(size_t)255;
        return pp;
    };
    float*  embp = (float*) carve((size_t)NB * CIN * HP_E * HP_E * 4);   // 4.46 MB
    float*  xp   = (float*) carve((size_t)NB * C2C * HP_M * HP_M * 4);   // 1.97 MB
    float*  refp = (float*) carve((size_t)NB * C2C * HP_R * HP_R * 4);   // 0.25 MB
    half_t* wnrm = (half_t*)carve((size_t)NB * LL * K1 * 2);             // 0.92 MB
    half_t* bcol = (half_t*)carve((size_t)NB * HWX * K1 * 2);            // 8.29 MB
    float*  yraw = (float*) carve((size_t)NB * LL * HWX * 4);            // 23.0 MB
    half_t* ypad = (half_t*)carve((size_t)NB * HP_M * HP_M * LPAD * 2);  // 12.8 MB
    half_t* rawT = (half_t*)carve((size_t)NB * 81 * CIN * LPAD * 2);     // 17.3 MB
    (void)in_sizes; (void)n_in; (void)out_size; (void)ws_size;

    const int TB = 256;

    // Stage 1: projections (zero-padded)
    {
        int total = NB * CIN * HP_E * HP_E;
        k_conv1x1_pad<<<(total + TB - 1) / TB, TB, 0, stream>>>(x, Wa, ba, aa, embp, CIN, 3, HP_E);
    }
    {
        int total = NB * C2C * HP_M * HP_M;
        k_conv1x1_pad<<<(total + TB - 1) / TB, TB, 0, stream>>>(x, Wm1, bm1, am1, xp, C2C, 1, HP_M);
    }
    {
        int total = NB * C2C * HP_R * HP_R;
        k_ref_pad<<<(total + TB - 1) / TB, TB, 0, stream>>>(x, Wm2, bm2, am2, refp);
    }

    // Stage 2: GEMM operand construction
    k_patchnorm<<<NB * LL, 128, 0, stream>>>(refp, wnrm);
    {
        int total = NB * HWX * K1;
        k_im2col<<<(total + TB - 1) / TB, TB, 0, stream>>>(xp, bcol);
    }
    {
        int total = NB * 81 * CIN * LPAD;
        k_rawT<<<(total + TB - 1) / TB, TB, 0, stream>>>(embp, rawT);
    }

    // Stage 3: correlation WMMA GEMM (2x2 blocked)
    {
        int waves = NB * MG_C * NG_X;               // 5876
        k_gemm_corr<<<(waves + 7) / 8, TB, 0, stream>>>(wnrm, bcol, yraw);
    }

    // Stage 4: softmax into zero-padded l-fastest layout
    {
        int n32 = (NB * HP_M * HP_M * LPAD) / 2;
        k_zero32<<<(n32 + TB - 1) / TB, TB, 0, stream>>>((uint32_t*)ypad, n32);
        int total = NB * HWX;
        k_softmax<<<(total + TB - 1) / TB, TB, 0, stream>>>(yraw, ypad);
    }

    // Stage 5: transposed-conv WMMA GEMMs (9 phases x 9 shifts, 2x2 blocked)
    {
        int waves = NB * 9 * 2 * NG_X;              // 8136
        k_gemm_tconv<<<(waves + 7) / 8, TB, 0, stream>>>(rawT, ypad, out);
    }
}
